// QuantizedLayer_55972013802094
// MI455X (gfx1250) — matrix operand order repaired and checked by reading, not compiled, not run on hardware
//
#include <hip/hip_runtime.h>

// ---------------------------------------------------------------------------
// QuantizedLayer: out[b,s,o] = sum_i x[b,s,i] * table[widx[o,i]] + btab[bidx[o]]
//   M = 4*2048 = 8192, K = 4096, N = 4096  ->  275 GFLOP GEMM (compute bound)
// Strategy:
//   1) dequant W -> f16 [N][K]   (ws)     2) convert x -> f16 [M][K] (ws)
//   3) WMMA f16 GEMM, f32 accumulate, bias fused in epilogue.
//      Wave tile 64x64 = 4x4 grid of 16x16 accumulators -> 16 WMMA : 16 b128
//      loads per K-step (1.0 loads/WMMA, ~32.8 FLOP per L1 byte).
// Whole f16 working set (96 MiB) fits in the 192 MB L2.
// ---------------------------------------------------------------------------

typedef __attribute__((ext_vector_type(16))) _Float16     v16h;
typedef __attribute__((ext_vector_type(8)))  float        v8f;
typedef __attribute__((ext_vector_type(4)))  unsigned int v4u;
typedef __attribute__((ext_vector_type(4)))  int          v4i;
typedef __attribute__((ext_vector_type(4)))  float        v4f;
typedef __attribute__((ext_vector_type(4)))  _Float16     v4h;

union Frag16 { v4u u[2]; v16h h; };

constexpr int K_DIM = 4096;   // IN_F
constexpr int N_DIM = 4096;   // OUT_F
constexpr int M_DIM = 4 * 2048;

// ---- W dequant: W16[o*K+i] = (f16) table[idx[o*K+i]], 4 elems per thread ----
__global__ void dequant_w_kernel(const int* __restrict__ idx,
                                 const float* __restrict__ table,
                                 _Float16* __restrict__ w16, int n4) {
  int i = blockIdx.x * blockDim.x + threadIdx.x;
  if (i >= n4) return;
  v4i v = ((const v4i*)idx)[i];
  v4h h;
  h.x = (_Float16)table[v.x];
  h.y = (_Float16)table[v.y];
  h.z = (_Float16)table[v.z];
  h.w = (_Float16)table[v.w];
  ((v4h*)w16)[i] = h;
}

// ---- x convert: f32 -> f16, 4 elems per thread ----
__global__ void cvt_x_kernel(const float* __restrict__ x,
                             _Float16* __restrict__ x16, int n4) {
  int i = blockIdx.x * blockDim.x + threadIdx.x;
  if (i >= n4) return;
  v4f v = ((const v4f*)x)[i];
  v4h h;
  h.x = (_Float16)v.x;
  h.y = (_Float16)v.y;
  h.z = (_Float16)v.z;
  h.w = (_Float16)v.w;
  ((v4h*)x16)[i] = h;
}

// ---- WMMA GEMM: block = 128x256 tile, 8 waves (2Mx4N), wave = 64x64 -------
__global__ __launch_bounds__(256)
void qgemm_wmma_kernel(const _Float16* __restrict__ X16,   // [M][K] row-major
                       const _Float16* __restrict__ W16,   // [N][K] row-major
                       const int*   __restrict__ bias_idx, // [N]
                       const float* __restrict__ bias_table,
                       float* __restrict__ out) {           // [M][N]
  constexpr int K = K_DIM, N = N_DIM;

  const int lane  = threadIdx.x & 31;
  const int wave  = threadIdx.x >> 5;
  const int hi    = lane >> 4;      // lane half: selects K sub-block per ISA layout
  const int mr    = lane & 15;      // row (A) / col (B) index within 16

  const int wave_m = wave >> 2;     // 0..1  -> 64 rows each
  const int wave_n = wave & 3;      // 0..3  -> 64 cols each
  const int mb0 = blockIdx.y * 128 + wave_m * 64;
  const int nb0 = blockIdx.x * 256 + wave_n * 64;

  // A fragment (16-bit A 16x32 layout): lane<16 holds K {kb..kb+7, kb+16..kb+23},
  // lane>=16 holds K {kb+8..kb+15, kb+24..kb+31} for row M = mr.
  const _Float16* pa[4];
#pragma unroll
  for (int i = 0; i < 4; ++i)
    pa[i] = X16 + (size_t)(mb0 + i * 16 + mr) * K + hi * 8;

  // B fragment (16-bit B 32x16 layout): lane holds column N = mr,
  // K {kb+hi*8..+7, kb+16+hi*8..+7} (same order as A).
  const _Float16* pb[4];
#pragma unroll
  for (int j = 0; j < 4; ++j)
    pb[j] = W16 + (size_t)(nb0 + j * 16 + mr) * K + hi * 8;

  v8f acc[4][4] = {};

  for (int kb = 0; kb < K; kb += 32) {
    Frag16 A[4], B[4];
#pragma unroll
    for (int i = 0; i < 4; ++i) {
      A[i].u[0] = *(const v4u*)(pa[i] + kb);        // K sub-block 0 (8 halves)
      A[i].u[1] = *(const v4u*)(pa[i] + kb + 16);   // K sub-block 1
    }
#pragma unroll
    for (int j = 0; j < 4; ++j) {
      B[j].u[0] = *(const v4u*)(pb[j] + kb);        // K sub-block 0 (8 halves)
      B[j].u[1] = *(const v4u*)(pb[j] + kb + 16);
    }
#pragma unroll
    for (int i = 0; i < 4; ++i)
#pragma unroll
      for (int j = 0; j < 4; ++j)
        acc[i][j] = __builtin_amdgcn_wmma_f32_16x16x32_f16(
            /*neg_a=*/false, A[i].h, /*neg_b=*/false, B[j].h,
            /*c_mod=*/(short)0, acc[i][j],
            /*reuse_a=*/false, /*reuse_b=*/false);
  }

  // Epilogue: C/D f32 layout -> VGPR v holds row M = mb + hi*8 + v, col N = mr.
#pragma unroll
  for (int j = 0; j < 4; ++j) {
    const int n = nb0 + j * 16 + mr;
    const float bias = bias_table[bias_idx[n]];
#pragma unroll
    for (int i = 0; i < 4; ++i) {
      const int mBase = mb0 + i * 16 + hi * 8;
#pragma unroll
      for (int v = 0; v < 8; ++v)
        out[(size_t)(mBase + v) * N + n] = acc[i][j][v] + bias;
    }
  }
}

extern "C" void kernel_launch(void* const* d_in, const int* in_sizes, int n_in,
                              void* d_out, int out_size, void* d_ws, size_t ws_size,
                              hipStream_t stream) {
  const float* x    = (const float*)d_in[0];   // [4,2048,4096] f32
  const int*   widx = (const int*)  d_in[1];   // [4096,4096]   i32
  const float* wtab = (const float*)d_in[2];   // [256]         f32
  const int*   bidx = (const int*)  d_in[3];   // [4096]        i32
  const float* btab = (const float*)d_in[4];   // [256]         f32
  float* out = (float*)d_out;                  // [8192,4096]   f32

  _Float16* X16 = (_Float16*)d_ws;                                       // 64 MiB
  _Float16* W16 = (_Float16*)((char*)d_ws + (size_t)M_DIM * K_DIM * 2);  // 32 MiB

  {
    int n4 = (M_DIM * K_DIM) / 4;
    cvt_x_kernel<<<(n4 + 255) / 256, 256, 0, stream>>>(x, X16, n4);
  }
  {
    int n4 = (N_DIM * K_DIM) / 4;
    dequant_w_kernel<<<(n4 + 255) / 256, 256, 0, stream>>>(widx, wtab, W16, n4);
  }

  dim3 grid(N_DIM / 256, M_DIM / 128);   // 16 x 64 blocks
  qgemm_wmma_kernel<<<grid, 256, 0, stream>>>(X16, W16, bidx, btab, out);
}
